// ModelDebug_18717467476434
// MI455X (gfx1250) — matrix operand order repaired and checked by reading, not compiled, not run on hardware
//
#include <hip/hip_runtime.h>
#include <stddef.h>

// ---------------------------------------------------------------------------
// Fused causal ReLU-attention for MI455X (gfx1250, wave32, WMMA 16x16x32 f16).
//
// B=8, T=1024, C=768, H=12, hd=64.
// FLOPs ~42G vs ~150MB HBM traffic -> near the compute/bandwidth crossover at
// 23.3 TB/s, so all matmuls run on v_wmma_f32_16x16x32_f16 (f32 accumulate);
// the ~54MB of f16 staging lives in the 192MB L2 between kernels.
//
// Layout trick: stage 1 stores v TRANSPOSED ([b,h,d,t]) so the attention
// kernel's second GEMM reads B-fragments K-contiguously from global (no LDS
// transpose, no block barriers), and the v-epilogue stores are contiguous
// 16-byte b128 stores.  All hot-loop addresses are one per-lane base pointer
// plus compile-time immediate offsets.
// ---------------------------------------------------------------------------

typedef __attribute__((ext_vector_type(16))) _Float16 v16h;
typedef __attribute__((ext_vector_type(8)))  _Float16 v8h;
typedef __attribute__((ext_vector_type(4)))  _Float16 v4h;
typedef __attribute__((ext_vector_type(8)))  float    v8f;
typedef __attribute__((ext_vector_type(4)))  float    v4f;

#define NB    8
#define NT    1024
#define NC    768
#define NH    12
#define HD    64
#define M_ALL (NB * NT)                     // 8192
#define N_ALL (3 * NC)                      // 2304
#define PLANE (NB * NH * NT * HD)           // 6291456 elems per q/k/v plane

__device__ inline v8f wmma_f16(v16h a, v16h b, v8f c) {
  return __builtin_amdgcn_wmma_f32_16x16x32_f16(false, a, false, b,
                                                (short)0, c, false, false);
}

__device__ inline v8h ld8(const _Float16* p) { return *(const v8h*)p; }

__device__ inline v16h make_frag(v8h lo, v8h hi) {
  v16h r;
#pragma unroll
  for (int i = 0; i < 8; ++i) { r[i] = lo[i]; r[i + 8] = hi[i]; }
  return r;
}

// Fragment lane bases (contiguous k = 16*(lane>>4) + e, f16, K=32):
//  A (16x32 row-major, ld):  base + (lane&15)*ld + 16*(lane>>4); 16B at +0,+8
//  B (32x16, K-contig, ld):  base + (lane&15)*ld + 16*(lane>>4); 16B at +0,+8

// ---------------------------------------------------------------------------
// Stage 0: f32 -> f16 staging of x and W.
// ---------------------------------------------------------------------------
__global__ void __launch_bounds__(256)
cvt_f32_to_f16(const float* __restrict__ in, _Float16* __restrict__ out, int n) {
  int i = (blockIdx.x * blockDim.x + threadIdx.x) * 4;
  if (i < n) {
    v4f d = *(const v4f*)(in + i);
    v4h o;
#pragma unroll
    for (int e = 0; e < 4; ++e) o[e] = (_Float16)d[e];
    *(v4h*)(out + i) = o;
  }
}

// ---------------------------------------------------------------------------
// Stage 1: QKV GEMM.  qkv[m,o] = sum_k x[m,k]*W[o,k] + b[o].
// Block = 256 threads = 8 waves -> 128x128 tile (wave = 32x64, 2x4 WMMA tiles).
// q,k stored [bh][t][d]; v stored TRANSPOSED [bh][d][t].
// ---------------------------------------------------------------------------
__global__ void __launch_bounds__(256)
qkv_gemm(const _Float16* __restrict__ xh, const _Float16* __restrict__ wh,
         const float* __restrict__ bias, _Float16* __restrict__ qkv) {
  const int lane = threadIdx.x & 31;
  const int wave = threadIdx.x >> 5;
  const int l = lane & 15, s = (lane >> 4) & 1;
  const int m0 = blockIdx.y * 128 + (wave >> 1) * 32;  // 4 waves down
  const int n0 = blockIdx.x * 128 + (wave & 1) * 64;   // 2 waves across

  // Per-lane base pointers; every offset below is a compile-time immediate.
  const _Float16* aP = xh + (size_t)(m0 + l) * NC + 16 * s;
  const _Float16* bP = wh + (size_t)(n0 + l) * NC + 16 * s;

  v8f acc[2][4] = {};
  for (int k0 = 0; k0 < NC; k0 += 32) {
    __builtin_prefetch(aP + 64, 0, 3);   // speculative global_prefetch_b8
    v16h a0 = make_frag(ld8(aP), ld8(aP + 8));
    v16h a1 = make_frag(ld8(aP + 16 * NC), ld8(aP + 16 * NC + 8));
    v16h b0 = make_frag(ld8(bP), ld8(bP + 8));
    v16h b1 = make_frag(ld8(bP + 16 * NC), ld8(bP + 16 * NC + 8));
    v16h b2 = make_frag(ld8(bP + 32 * NC), ld8(bP + 32 * NC + 8));
    v16h b3 = make_frag(ld8(bP + 48 * NC), ld8(bP + 48 * NC + 8));
    acc[0][0] = wmma_f16(a0, b0, acc[0][0]);
    acc[1][0] = wmma_f16(a1, b0, acc[1][0]);
    acc[0][1] = wmma_f16(a0, b1, acc[0][1]);
    acc[1][1] = wmma_f16(a1, b1, acc[1][1]);
    acc[0][2] = wmma_f16(a0, b2, acc[0][2]);
    acc[1][2] = wmma_f16(a1, b2, acc[1][2]);
    acc[0][3] = wmma_f16(a0, b3, acc[0][3]);
    acc[1][3] = wmma_f16(a1, b3, acc[1][3]);
    aP += 32;
    bP += 32;
  }

  // Epilogue.  A 64-wide wave tile never straddles the q/k/v boundaries
  // (768 and 1536 are 64-aligned), so `which` is wave-uniform: decide it once
  // via readfirstlane and branch scalar -- no exec-mask divergence.
  const int n0u = __builtin_amdgcn_readfirstlane(n0);
  const int which = (n0u >= 2 * NC) ? 2 : ((n0u >= NC) ? 1 : 0);
  // A 128-row block tile never crosses a batch boundary (1024 % 128 == 0).
  const unsigned biOff = (unsigned)(m0 >> 10) * (unsigned)(NH * NT * HD);
  const int tb = (m0 & (NT - 1)) + 8 * s;   // t of accumulator row r=0, i=0

  if (which == 2) {
    // v: [bh][d][t] -- 8 accumulator rows are contiguous in t => b128 stores.
#pragma unroll
    for (int j = 0; j < 4; ++j) {
      const int n = n0 + 16 * j + l;
      const float bv = bias[n];
      const int c = n - 2 * NC;
      const int h = c >> 6, d = c & 63;
      _Float16* dst = qkv + (2u * PLANE + biOff + (unsigned)h * (HD * NT)
                             + (unsigned)d * NT + (unsigned)tb);
#pragma unroll
      for (int i = 0; i < 2; ++i) {
        v8h pk;
#pragma unroll
        for (int r = 0; r < 8; ++r) pk[r] = (_Float16)(acc[i][j][r] + bv);
        *(v8h*)(dst + 16 * i) = pk;
      }
    }
  } else {
    // q,k: [bh][t][d] -- base pointer once, immediate offsets (16i+r)*HD.
#pragma unroll
    for (int j = 0; j < 4; ++j) {
      const int n = n0 + 16 * j + l;
      const float bv = bias[n];
      const int c = n - which * NC;
      const int h = c >> 6, d = c & 63;
      _Float16* dst = qkv + ((unsigned)which * PLANE + biOff
                             + (unsigned)h * (NT * HD)
                             + (unsigned)tb * HD + (unsigned)d);
#pragma unroll
      for (int i = 0; i < 2; ++i)
#pragma unroll
        for (int r = 0; r < 8; ++r)
          dst[(16 * i + r) * HD] = (_Float16)(acc[i][j][r] + bv);
    }
  }
}

// ---------------------------------------------------------------------------
// Stage 2: per-(b,h) causal ReLU attention.  Plain accumulation over causal
// k-tiles (masked ReLU, no softmax).  Block = 128 threads = 4 waves; each
// wave owns 16 q rows and runs independently (no block barriers).  Only LDS
// use: the wave-private S-tile D-layout -> A-layout round trip.
// ---------------------------------------------------------------------------
__global__ void __launch_bounds__(128)
attn_relu(const _Float16* __restrict__ qkv, float* __restrict__ out) {
  __shared__ _Float16 sS[4][16][72];     // per-wave S strip (16 x 64), padded

  const int lane = threadIdx.x & 31;
  const int wave = threadIdx.x >> 5;
  const int l = lane & 15, s = (lane >> 4) & 1;
  const int bh = blockIdx.y;             // b*NH + h
  const int b = bh / NH, h = bh - b * NH;
  const int qt = blockIdx.x;             // 64-row q tile index
  const int qrow0 = qt * 64 + wave * 16;

  // q fragments: loaded once, reused across every k tile.
  v16h qa0, qa1;
  {
    const _Float16* qa = qkv + (size_t)bh * NT * HD + (size_t)(qrow0 + l) * HD + 16 * s;
    qa0 = make_frag(ld8(qa), ld8(qa + 8));
    qa1 = make_frag(ld8(qa + 32), ld8(qa + 40));
  }
  // Per-lane base pointers (B-fragment pattern), advanced per k tile.
  const _Float16* kP = qkv + 1u * PLANE + (size_t)bh * NT * HD + (size_t)l * HD + 16 * s;
  const _Float16* vP = qkv + 2u * PLANE + (size_t)bh * HD * NT + (size_t)l * NT + 16 * s;
  // Wave-private LDS S tile: store base (D layout) and load base (A layout).
  _Float16* sSt = &sS[wave][8 * s][l];          // + r*72 + nt*16
  const _Float16* sLd = &sS[wave][l][16 * s];   // + kk0, +8

  v8f yacc[4] = {};
  const int nkt = qt + 1;                // causal: only k tiles <= q tile
  const int m_g0 = qrow0 + 8 * s;

  for (int kt = 0; kt < nkt; ++kt) {
    // S = relu(causal_mask(q . k^T)) * scale, into wave-private LDS (f16).
#pragma unroll
    for (int nt = 0; nt < 4; ++nt) {
      v16h kb0 = make_frag(ld8(kP + nt * 16 * HD), ld8(kP + nt * 16 * HD + 8));
      v16h kb1 = make_frag(ld8(kP + nt * 16 * HD + 32), ld8(kP + nt * 16 * HD + 40));
      v8f sacc = {};
      sacc = wmma_f16(qa0, kb0, sacc);
      sacc = wmma_f16(qa1, kb1, sacc);
      const int n_g = kt * 64 + nt * 16 + l;
#pragma unroll
      for (int r = 0; r < 8; ++r) {
        float v = sacc[r] * 0.125f;                        // 1/sqrt(64)
        v = (n_g <= m_g0 + r) ? fmaxf(v, 0.0f) : 0.0f;     // causal + ReLU
        sSt[r * 72 + nt * 16] = (_Float16)v;
      }
    }
    __builtin_amdgcn_wave_barrier();   // scheduling fence (wave-private LDS)

    // Y += S(16x64) @ V(64x64): S from LDS as A-frags; v B-frags straight
    // from global in the transposed [d][t] layout (t is the K dimension).
#pragma unroll
    for (int kk0 = 0; kk0 < 64; kk0 += 32) {
      v16h sa = make_frag(*(const v8h*)(sLd + kk0), *(const v8h*)(sLd + kk0 + 8));
#pragma unroll
      for (int nt = 0; nt < 4; ++nt) {
        v16h vb = make_frag(ld8(vP + nt * 16 * NT + kk0),
                            ld8(vP + nt * 16 * NT + kk0 + 8));
        yacc[nt] = wmma_f16(sa, vb, yacc[nt]);
      }
    }
    __builtin_amdgcn_wave_barrier();

    kP += 64 * HD;   // next 64 k rows
    vP += 64;        // next 64 t columns
  }

  // Store y strip: base pointer once, immediate offsets r*NC + nt*16.
  float* ob = out + ((size_t)b * NT + m_g0) * NC + h * HD + l;
#pragma unroll
  for (int nt = 0; nt < 4; ++nt)
#pragma unroll
    for (int r = 0; r < 8; ++r)
      ob[r * NC + nt * 16] = yacc[nt][r];
}

// ---------------------------------------------------------------------------
extern "C" void kernel_launch(void* const* d_in, const int* in_sizes, int n_in,
                              void* d_out, int out_size, void* d_ws, size_t ws_size,
                              hipStream_t stream) {
  const float* x    = (const float*)d_in[0];   // [8,1024,768]
  const float* W    = (const float*)d_in[1];   // [2304,768]
  const float* bias = (const float*)d_in[2];   // [2304]
  float* out = (float*)d_out;                  // [8,1024,768]

  char* ws = (char*)d_ws;
  const size_t n_x = (size_t)M_ALL * NC;       // 6291456
  const size_t n_w = (size_t)N_ALL * NC;       // 1769472
  _Float16* xh   = (_Float16*)ws;
  _Float16* wh   = (_Float16*)(ws + n_x * sizeof(_Float16));
  _Float16* qkvh = (_Float16*)(ws + (n_x + n_w) * sizeof(_Float16));

  cvt_f32_to_f16<<<(int)(n_x / 4 / 256), 256, 0, stream>>>(x, xh, (int)n_x);
  cvt_f32_to_f16<<<(int)(n_w / 4 / 256), 256, 0, stream>>>(W, wh, (int)n_w);

  // QKV GEMM: grid (N/128, M/128) = (18, 64), 256 threads.
  qkv_gemm<<<dim3(N_ALL / 128, M_ALL / 128), 256, 0, stream>>>(xh, wh, bias, qkvh);

  // Attention: grid (T/64 q-tiles, B*H) = (16, 96), 128 threads.
  attn_relu<<<dim3(NT / 64, NB * NH), 128, 0, stream>>>(qkvh, out);
}
